// EmitterVectorQuantizer_32195074851355
// MI455X (gfx1250) — matrix operand order repaired and checked by reading, not compiled, not run on hardware
//
#include <hip/hip_runtime.h>
#include <hip/hip_bf16.h>

// ---------------------------------------------------------------------------
// VQ-VAE vector quantizer for MI455X (gfx1250, wave32).
//   f:[8,256,80,80] f32, E:[8192,256] f32
//   score(n,k) = ||e_k||^2 - 2 f_n.e_k   (||f||^2 dropped: constant per row)
//   GEMM: bf16x3 split WMMA (v_wmma_f32_16x16x32_bf16), ~f32 accuracy.
//   B tiles streamed into LDS by the Tensor Data Mover (tensor_load_to_lds),
//   double-buffered on TENSORcnt with one barrier per tile. The TDM pad
//   feature (12 DWORDs per 128) lands rows at stride 280 bf16 -> 560B ->
//   bank-conflict-free ds_load_b128 fragment reads.
// ---------------------------------------------------------------------------

typedef __attribute__((ext_vector_type(16))) __bf16 v16bf;
typedef __attribute__((ext_vector_type(8)))  float  v8f;
typedef __attribute__((ext_vector_type(4)))  unsigned int u32x4;
typedef __attribute__((ext_vector_type(8)))  int    i32x8;
typedef __attribute__((ext_vector_type(4)))  int    i32x4;

#define C_DIM   256
#define HW_DIM  6400          // 80*80
#define N_ROWS  51200         // B*HW
#define K_CODES 8192
#define NUMEL   13107200      // B*C*HW
#define M_TILE  128           // rows per workgroup (8 waves x 16)
#define N_TILES (K_CODES / 16)   // 512 code tiles
#define SEG     280           // bf16 elems per LDS row: 256 + 24 pad (TDM pad)

#if defined(__has_builtin)
#  if __has_builtin(__builtin_amdgcn_tensor_load_to_lds)
#    define HAVE_TDM 1
#  endif
#endif
#ifndef HAVE_TDM
#  define HAVE_TDM 0
#endif

union BF16x16 { v16bf v; unsigned int u[8]; uint4 q[2]; };

__device__ __forceinline__ unsigned short bf_hi_rn(float x) {
  unsigned int u = __float_as_uint(x);
  unsigned int r = u + 0x7FFFu + ((u >> 16) & 1u);
  return (unsigned short)(r >> 16);
}

__device__ __forceinline__ void bf_split(float x, unsigned short& h, unsigned short& l) {
  h = bf_hi_rn(x);
  float hf = __uint_as_float(((unsigned int)h) << 16);
  l = bf_hi_rn(x - hf);
}

__device__ __forceinline__ v8f wmma_bf16(v16bf a, v16bf b, v8f c) {
  return __builtin_amdgcn_wmma_f32_16x16x32_bf16(
      false, a, false, b, (short)0, c, false, false);
}

#if HAVE_TDM
// Issue one TDM copy: 16 rows x 256 bf16 (one plane of one code tile) from
// global into LDS, padding 12 DWORDs after every 128 DWORDs (one 256-elem
// row) so LDS rows land with stride SEG=280 bf16.
__device__ __forceinline__ void tdm_issue_tile(const unsigned short* plane,
                                               int tile, unsigned lds_byte_addr) {
  unsigned long long ga =
      (unsigned long long)(size_t)plane + (unsigned long long)tile * (16u * C_DIM * 2u);
  u32x4 g0;
  g0.x = 1u;                                   // count=1, user mode, no gather
  g0.y = lds_byte_addr;                        // LDS destination
  g0.z = (unsigned)(ga & 0xffffffffu);         // global_addr[31:0]
  g0.w = (unsigned)((ga >> 32) & 0x01ffffffu)  // global_addr[56:32]
         | 0x80000000u;                        // type=2 ("image") in [127:126]
  i32x8 g1;
  g1[0] = (int)((1u << 16)     // data_size = 1 -> 2 bytes
              | (1u << 20)     // pad_enable
              | (6u << 22)     // pad_interval: 128 DWORDs (one 256-elem row)
              | (11u << 25));  // pad_amount: 12 DWORDs (24 bf16)
  g1[1] = 0x01000000;          // tensor_dim0 = 256 (bits 79:48 -> [31:16])
  g1[2] = 0x20000000;          // tensor_dim1 = 8192 (bits 111:80 -> [31:16])
  g1[3] = 0x01000000;          // tile_dim0 = 256 (bits 127:112)
  g1[4] = 16;                  // tile_dim1 = 16 rows
  g1[5] = C_DIM;               // tensor_dim0_stride = 256
  g1[6] = 0;
  g1[7] = 0;
  i32x4 z4 = {0, 0, 0, 0};
#if __clang_major__ >= 23
  i32x8 z8 = {0, 0, 0, 0, 0, 0, 0, 0};
  __builtin_amdgcn_tensor_load_to_lds(g0, g1, z4, z4, z8, 0);
#else
  __builtin_amdgcn_tensor_load_to_lds(g0, g1, z4, z4, 0);
#endif
}
#endif

// ---------------------------------------------------------------------------
// Kernel 1: per-code squared norms + bf16 hi/lo split of codebook.
// ---------------------------------------------------------------------------
__global__ __launch_bounds__(256)
void vq_prep(const float* __restrict__ embed, float* __restrict__ e_norm,
             unsigned short* __restrict__ e_hi, unsigned short* __restrict__ e_lo,
             float* __restrict__ loss_acc) {
  if (blockIdx.x == 0 && threadIdx.x == 0) *loss_acc = 0.0f;
  const int wave = threadIdx.x >> 5;
  const int lane = threadIdx.x & 31;
  const int k = blockIdx.x * 8 + wave;
  const float* row = embed + (size_t)k * C_DIM;
  float sum = 0.0f;
#pragma unroll
  for (int i = 0; i < 8; ++i) {
    const int c = lane + 32 * i;
    const float x = row[c];
    unsigned short h, l;
    bf_split(x, h, l);
    e_hi[(size_t)k * C_DIM + c] = h;
    e_lo[(size_t)k * C_DIM + c] = l;
    sum += x * x;
  }
#pragma unroll
  for (int m = 16; m >= 1; m >>= 1) sum += __shfl_xor(sum, m, 32);
  if (lane == 0) e_norm[k] = sum;
}

// ---------------------------------------------------------------------------
// Kernel 2: tiled GEMM (bf16x3 WMMA) + running argmin.
// 400 blocks x 256 threads. Wave w owns rows [blk*128 + w*16, +16).
// A fragments (hi+lo, all 8 C-chunks) persist in registers.
// B tiles DMA'd into LDS by the TDM (leader wave only, scalar branch),
// double-buffered on TENSORcnt with one barrier per tile.
// ---------------------------------------------------------------------------
__global__ __launch_bounds__(256)
void vq_gemm_argmin(const float* __restrict__ f,
                    const unsigned short* __restrict__ e_hi,
                    const unsigned short* __restrict__ e_lo,
                    const float* __restrict__ e_norm,
                    int* __restrict__ best_idx) {
  __shared__ __align__(16) unsigned short ldsHi[2][16 * SEG];
  __shared__ __align__(16) unsigned short ldsLo[2][16 * SEG];

  const int tid   = threadIdx.x;
  const int wave  = tid >> 5;
  const int lane  = tid & 31;
  const int rr    = lane & 15;       // A: row-in-tile / B & D: code-in-tile
  const int khalf = lane >> 4;       // k-half selector within 32-wide chunk
  const int row_base = blockIdx.x * M_TILE + wave * 16;

  // Scalar (SGPR) leader predicate: forces s_cmp + s_cbranch, so only wave 0
  // issues TDM descriptors / waits TENSORcnt (tensor ops ignore EXEC).
  const bool leader = ((__builtin_amdgcn_readfirstlane(tid) >> 5) == 0);

  // ---- Load + split the A tile (16 rows x 256 C) into registers ----------
  const int n_glob = row_base + rr;
  const int b_idx  = n_glob / HW_DIM;
  const int hw     = n_glob - b_idx * HW_DIM;
  const float* fb = f + (size_t)b_idx * C_DIM * HW_DIM + hw;
  const int kh8 = khalf * 8;

  BF16x16 ah[8], al[8];
#pragma unroll
  for (int cc = 0; cc < 8; ++cc) {
#pragma unroll
    for (int j = 0; j < 8; ++j) {
      const int cbase = cc * 32 + 2 * j + 2 * kh8;
      const float x0 = fb[(size_t)cbase * HW_DIM];
      const float x1 = fb[(size_t)(cbase + 1) * HW_DIM];
      unsigned short h0, l0, h1, l1;
      bf_split(x0, h0, l0);
      bf_split(x1, h1, l1);
      ah[cc].u[j] = (unsigned int)h0 | ((unsigned int)h1 << 16);
      al[cc].u[j] = (unsigned int)l0 | ((unsigned int)l1 << 16);
    }
  }

  float rmin[8];
  int   ridx[8];
#pragma unroll
  for (int r = 0; r < 8; ++r) { rmin[r] = 3.4e38f; ridx[r] = 0; }

#if HAVE_TDM
  // Prologue: DMA tile 0 (hi + lo planes) into buffer 0.
  if (leader) {
    tdm_issue_tile(e_hi, 0, (unsigned)(size_t)&ldsHi[0][0]);
    tdm_issue_tile(e_lo, 0, (unsigned)(size_t)&ldsLo[0][0]);
  }
#endif

  for (int t = 0; t < N_TILES; ++t) {
    const int buf = t & 1;

#if HAVE_TDM
    if (leader) __builtin_amdgcn_s_wait_tensorcnt(0);      // tile t landed
    __syncthreads();   // tile t visible; everyone done reading buf^1 (iter t-1)
    if (leader && t + 1 < N_TILES) {
      tdm_issue_tile(e_hi, t + 1, (unsigned)(size_t)&ldsHi[buf ^ 1][0]);
      tdm_issue_tile(e_lo, t + 1, (unsigned)(size_t)&ldsLo[buf ^ 1][0]);
    }
#else
    // Fallback: cooperative copy (no TDM builtin on this toolchain).
    __syncthreads();
    {
      const size_t c0 = (size_t)t * 16 * C_DIM;
#pragma unroll
      for (int i = 0; i < 2; ++i) {
        const int q = tid + i * 256, row = q >> 5, col = q & 31;
        *(uint4*)&ldsHi[buf][row * SEG + col * 8] =
            *(const uint4*)(e_hi + c0 + (size_t)row * C_DIM + col * 8);
        *(uint4*)&ldsLo[buf][row * SEG + col * 8] =
            *(const uint4*)(e_lo + c0 + (size_t)row * C_DIM + col * 8);
      }
    }
    __syncthreads();
#endif

    // Three independent accumulator chains: hi*hi, hi*lo, lo*hi.
    v8f a0 = 0.0f, a1 = 0.0f, a2 = 0.0f;
#pragma unroll
    for (int cc = 0; cc < 8; ++cc) {
      // B order used here (same as A): lane -> code n, dwords 0..7 -> 16 contiguous K-vals
      // (lanes 0-15: K 0..15, lanes 16-31: K 16..31 of this 32-chunk).
      const int off = rr * SEG + cc * 32 + khalf * 16;
      BF16x16 bh, bl;
      bh.q[0] = *(const uint4*)&ldsHi[buf][off];
      bh.q[1] = *(const uint4*)&ldsHi[buf][off + 8];
      bl.q[0] = *(const uint4*)&ldsLo[buf][off];
      bl.q[1] = *(const uint4*)&ldsLo[buf][off + 8];
      a0 = wmma_bf16(ah[cc].v, bh.v, a0);
      a1 = wmma_bf16(ah[cc].v, bl.v, a1);
      a2 = wmma_bf16(al[cc].v, bh.v, a2);
    }

    const int code = t * 16 + rr;
    const float en = e_norm[code];
#pragma unroll
    for (int r = 0; r < 8; ++r) {
      const float s = en - 2.0f * (a0[r] + a1[r] + a2[r]);
      const bool take = s < rmin[r];           // strict: keep earliest code
      rmin[r] = take ? s : rmin[r];
      ridx[r] = take ? code : ridx[r];
    }
  }

  // ---- Cross-lane argmin within each 16-lane half ------------------------
#pragma unroll
  for (int r = 0; r < 8; ++r) {
#pragma unroll
    for (int m = 8; m >= 1; m >>= 1) {
      const float s2 = __shfl_xor(rmin[r], m, 32);
      const int   i2 = __shfl_xor(ridx[r], m, 32);
      const bool take = (s2 < rmin[r]) || (s2 == rmin[r] && i2 < ridx[r]);
      rmin[r] = take ? s2 : rmin[r];
      ridx[r] = take ? i2 : ridx[r];
    }
  }
  if (rr == 0) {
    const int m0 = row_base + khalf * 8;   // lane0 -> rows 0..7, lane16 -> 8..15
#pragma unroll
    for (int r = 0; r < 8; ++r) best_idx[m0 + r] = ridx[r];
  }
}

// ---------------------------------------------------------------------------
// Kernel 3: gather quantized vectors (f32 codebook), write quantized_st
// (numerically == q), accumulate sum of (f - q)^2.
// ---------------------------------------------------------------------------
__global__ __launch_bounds__(256)
void vq_gather_loss(const float* __restrict__ f, const float* __restrict__ embed,
                    const int* __restrict__ best_idx, float* __restrict__ out_q,
                    float* __restrict__ loss_acc) {
  const size_t gid = (size_t)blockIdx.x * 256 + threadIdx.x;   // < NUMEL
  const int bi = (int)(gid / ((size_t)C_DIM * HW_DIM));
  const int c  = (int)((gid / HW_DIM) % C_DIM);
  const int hw = (int)(gid % HW_DIM);
  const int n  = bi * HW_DIM + hw;
  const int code = best_idx[n];
  const float q = embed[(size_t)code * C_DIM + c];
  const float x = f[gid];
  out_q[gid] = q;                        // f + sg(q - f) == q
  const float d = x - q;
  float p = d * d;
#pragma unroll
  for (int m = 16; m >= 1; m >>= 1) p += __shfl_xor(p, m, 32);
  __shared__ float red[8];
  const int wave = threadIdx.x >> 5, lane = threadIdx.x & 31;
  if (lane == 0) red[wave] = p;
  __syncthreads();
  if (wave == 0) {
    float t = (lane < 8) ? red[lane] : 0.0f;
#pragma unroll
    for (int m = 4; m >= 1; m >>= 1) t += __shfl_xor(t, m, 32);
    if (lane == 0) atomicAdd(loss_acc, t);
  }
}

// ---------------------------------------------------------------------------
// Kernel 4: final loss scalar + indices (as float) into d_out tail.
// ---------------------------------------------------------------------------
__global__ __launch_bounds__(256)
void vq_finalize(const int* __restrict__ best_idx,
                 const float* __restrict__ loss_acc, float* __restrict__ out) {
  const int gid = blockIdx.x * 256 + threadIdx.x;    // 200*256 = 51200
  if (gid < N_ROWS) out[(size_t)NUMEL + 1 + gid] = (float)best_idx[gid];
  if (gid == 0) {
    // vq_loss = embedding + BETA*commitment = (1 + 0.25) * mean((f-q)^2)
    out[NUMEL] = 1.25f * (*loss_acc) / (float)NUMEL;
  }
}

// ---------------------------------------------------------------------------
extern "C" void kernel_launch(void* const* d_in, const int* in_sizes, int n_in,
                              void* d_out, int out_size, void* d_ws, size_t ws_size,
                              hipStream_t stream) {
  const float* f     = (const float*)d_in[0];   // [8,256,80,80]
  const float* embed = (const float*)d_in[1];   // [8192,256]
  float* out = (float*)d_out;

  // Workspace layout (~8.6 MB)
  char* ws = (char*)d_ws;
  float*          e_norm  = (float*)ws;                                   // 32 KB
  unsigned short* e_hi    = (unsigned short*)(ws + 32768);                // 4 MB
  unsigned short* e_lo    = (unsigned short*)(ws + 32768 + 4194304);      // 4 MB
  int*            bestidx = (int*)(ws + 32768 + 2 * 4194304);             // 200 KB
  float*          lossacc = (float*)(ws + 32768 + 2 * 4194304 + 204800);  // 4 B

  vq_prep<<<K_CODES / 8, 256, 0, stream>>>(embed, e_norm, e_hi, e_lo, lossacc);
  vq_gemm_argmin<<<N_ROWS / M_TILE, 256, 0, stream>>>(f, e_hi, e_lo, e_norm, bestidx);
  vq_gather_loss<<<NUMEL / 256, 256, 0, stream>>>(f, embed, bestidx, out, lossacc);
  vq_finalize<<<N_ROWS / 256, 256, 0, stream>>>(bestidx, lossacc, out);
}
